// MSARowAttentionV47_42374147342902
// MI455X (gfx1250) — matrix operand order repaired and checked by reading, not compiled, not run on hardware
//
#include <hip/hip_runtime.h>
#include <hip/hip_bf16.h>

// ---------------------------------------------------------------------------
// Problem constants
// ---------------------------------------------------------------------------
#define B_  1
#define S_  128
#define N_  256
#define C_  256
#define H_  8
#define DH_ 32
#define PD_ 128
#define SN_ (S_ * N_)           // 32768 rows
#define SCALE_ 0.17677669529663687f   // 32^-0.5
#define NEG_   -1000000000.0f
#define EPS_   1e-5f

// ---------------------------------------------------------------------------
// Vector types for WMMA
// ---------------------------------------------------------------------------
typedef __attribute__((ext_vector_type(16))) unsigned short v16u;
typedef __attribute__((ext_vector_type(16))) __bf16         v16bf;
typedef __attribute__((ext_vector_type(8)))  float          v8f;

__device__ __forceinline__ v8f v8f_zero() {
  v8f z = {0.f, 0.f, 0.f, 0.f, 0.f, 0.f, 0.f, 0.f};
  return z;
}

__device__ __forceinline__ unsigned short f32_to_bf16(float f) {
  unsigned int u = __builtin_bit_cast(unsigned int, f);
  unsigned int lsb = (u >> 16) & 1u;
  u += 0x7fffu + lsb;               // round to nearest even
  return (unsigned short)(u >> 16);
}

__device__ __forceinline__ v8f wmma_bf16f32(v16u a, v16u b, v8f c) {
  return __builtin_amdgcn_wmma_f32_16x16x32_bf16(
      false, __builtin_bit_cast(v16bf, a),
      false, __builtin_bit_cast(v16bf, b),
      (short)0, c, false, false);
}

// ---------------------------------------------------------------------------
// CDNA5 async global->LDS copy (ASYNCcnt-tracked, no VGPR staging)
// ---------------------------------------------------------------------------
__device__ __forceinline__ void async_ld_b128(const void* lds_dst, const void* gsrc) {
  unsigned dst = (unsigned)(size_t)lds_dst;   // low 32 bits of LDS aperture = LDS byte offset
  asm volatile("global_load_async_to_lds_b128 %0, %1, off"
               :: "v"(dst), "v"(gsrc) : "memory");
}
__device__ __forceinline__ void wait_async0() {
  asm volatile("s_wait_asynccnt 0" ::: "memory");
}

// Stage a 128x32 bf16 tile (rows row0..row0+127, cols k0..k0+31 of a [*][256]
// row-major matrix) into LDS via 512 async b128 transfers (2 per thread @256).
__device__ __forceinline__ void async_tile_load(unsigned short* ldsbuf,
                                                const unsigned short* gbase,
                                                int row0, int k0, int tid) {
#pragma unroll
  for (int i = 0; i < 2; ++i) {
    int t = tid + i * 256;                 // 0..511
    int r = t >> 2, c = (t & 3) * 8;       // 4 x b128 per 32-elem row
    async_ld_b128(ldsbuf + t * 8, gbase + (size_t)(row0 + r) * 256 + k0 + c);
  }
}

// A fragment (16x32 bf16, M x K):
// lanes 0-15: row M = lane, K in {0..7, 16..23}; lanes 16-31: row M = lane-16,
// K in {8..15, 24..31}. Element i: k = kb + (i&7) + (i>=8 ? 16 : 0).
__device__ __forceinline__ v16u load_a_frag(const unsigned short* lds,
                                            int row0, int ld, int k0, int lane) {
  int m  = row0 + (lane & 15);
  int kb = k0 + ((lane & 16) ? 8 : 0);
  v16u a;
#pragma unroll
  for (int i = 0; i < 16; ++i) {
    int k = kb + (i & 7) + ((i & 8) << 1);
    a[i] = lds[m * ld + k];
  }
  return a;
}

// B fragment (32x16 bf16, K x N) where source is stored [n][k] (row n, stride ld).
__device__ __forceinline__ v16u load_b_frag_nk(const unsigned short* lds,
                                               int n0, int ld, int lane) {
  int n  = n0 + (lane & 15);
  int kb = (lane & 16) >> 1;
  v16u b;
#pragma unroll
  for (int i = 0; i < 16; ++i) b[i] = lds[n * ld + kb + (i & 7) + ((i & 8) << 1)];
  return b;
}

// B fragment (32x16) where source is stored [k][n] (row k, stride ld).
__device__ __forceinline__ v16u load_b_frag_kn(const unsigned short* lds,
                                               int k0, int n0, int ld, int lane) {
  int n  = n0 + (lane & 15);
  int kb = k0 + ((lane & 16) >> 1);
  v16u b;
#pragma unroll
  for (int i = 0; i < 16; ++i) b[i] = lds[(kb + (i & 7) + ((i & 8) << 1)) * ld + n];
  return b;
}

// ---------------------------------------------------------------------------
// K1: f32 -> bf16 conversion
// ---------------------------------------------------------------------------
__global__ __launch_bounds__(256) void f2bf_kernel(const float* __restrict__ src,
                                                   unsigned short* __restrict__ dst,
                                                   int n) {
  int i = blockIdx.x * 256 + threadIdx.x;
  if (i < n) dst[i] = f32_to_bf16(src[i]);
}

// ---------------------------------------------------------------------------
// K2: pair bias  bias[h][i][j] = sum_d pair[i][j][d] * pb_w[h][d] + pb_b[h]
// ---------------------------------------------------------------------------
__global__ __launch_bounds__(256) void pair_bias_kernel(const float* __restrict__ pair,
                                                        const float* __restrict__ pbw,
                                                        const float* __restrict__ pbb,
                                                        float* __restrict__ bias) {
  int ij = blockIdx.x * 256 + threadIdx.x;   // 0..65535 over (i,j)
  if (ij >= N_ * N_) return;
  const float* p = pair + (size_t)ij * PD_;
  float acc[H_];
#pragma unroll
  for (int h = 0; h < H_; ++h) acc[h] = 0.f;
  for (int d = 0; d < PD_; ++d) {
    float pv = p[d];
#pragma unroll
    for (int h = 0; h < H_; ++h) acc[h] += pv * pbw[h * PD_ + d];
  }
#pragma unroll
  for (int h = 0; h < H_; ++h)
    bias[(size_t)h * N_ * N_ + ij] = acc[h] + pbb[h];
}

// ---------------------------------------------------------------------------
// K3: QKV GEMM  out[row,o] = sum_k msa_bf[row,k] * Wqkv[o,k] + b[o]
//     block tile 128x128, K=256 in 8 steps of 32, double-buffered LDS filled
//     by async global->LDS DMA; 8 waves (4x2), wave = 32x64 (8 WMMA / k-step)
// ---------------------------------------------------------------------------
__global__ __launch_bounds__(256) void qkv_gemm_kernel(
    const unsigned short* __restrict__ A,   // [SN][256] bf16
    const unsigned short* __restrict__ W,   // [768][256] bf16
    const float* __restrict__ bvec,         // [768]
    unsigned short* __restrict__ qo,
    unsigned short* __restrict__ ko,
    unsigned short* __restrict__ vo) {
  __shared__ unsigned short at[2][128 * 32];
  __shared__ unsigned short bt[2][128 * 32];
  int tid = threadIdx.x, lane = tid & 31, w = tid >> 5;
  int row0 = blockIdx.x * 128;
  int col0 = blockIdx.y * 128;
  int wm = w >> 1, wn = w & 1;

  v8f acc[2][4];
#pragma unroll
  for (int mi = 0; mi < 2; ++mi)
#pragma unroll
    for (int j = 0; j < 4; ++j) acc[mi][j] = v8f_zero();

  async_tile_load(at[0], A, row0, 0, tid);
  async_tile_load(bt[0], W, col0, 0, tid);
  wait_async0();
  __syncthreads();

#pragma unroll
  for (int kk = 0; kk < 8; ++kk) {
    int cur = kk & 1;
    if (kk < 7) {                           // prefetch next tile into other buffer
      async_tile_load(at[cur ^ 1], A, row0, (kk + 1) * 32, tid);
      async_tile_load(bt[cur ^ 1], W, col0, (kk + 1) * 32, tid);
    }
    v16u a0 = load_a_frag(at[cur], wm * 32, 32, 0, lane);
    v16u a1 = load_a_frag(at[cur], wm * 32 + 16, 32, 0, lane);
#pragma unroll
    for (int j = 0; j < 4; ++j) {
      v16u b = load_b_frag_nk(bt[cur], wn * 64 + j * 16, 32, lane);
      acc[0][j] = wmma_bf16f32(a0, b, acc[0][j]);
      acc[1][j] = wmma_bf16f32(a1, b, acc[1][j]);
    }
    wait_async0();
    __syncthreads();
  }

  int hi = lane >> 4, nl = lane & 15;
#pragma unroll
  for (int mi = 0; mi < 2; ++mi)
#pragma unroll
    for (int j = 0; j < 4; ++j)
#pragma unroll
      for (int r = 0; r < 8; ++r) {
        int grow = row0 + wm * 32 + mi * 16 + r + 8 * hi;
        int gcol = col0 + wn * 64 + j * 16 + nl;
        float val = acc[mi][j][r] + bvec[gcol];
        int s = grow >> 8, n = grow & 255;
        int g = gcol >> 8, h = (gcol >> 5) & 7, dh = gcol & 31;
        unsigned short bb = f32_to_bf16(val);
        unsigned short* dst = (g == 0) ? qo : (g == 1) ? ko : vo;
        dst[(((size_t)(s * H_ + h)) * N_ + n) * DH_ + dh] = bb;
      }
}

// ---------------------------------------------------------------------------
// K4: attention per (s,h). grid.x = S*H, grid.y = 4 (64-row query slabs),
//     128 threads = 4 waves, each wave owns 16 query rows.
//     K/V staged into LDS via async DMA; scores live in registers
//     (16 v8f tiles / wave); softmax in-register via shfl_xor.
// ---------------------------------------------------------------------------
__global__ __launch_bounds__(128) void attn_kernel(
    const unsigned short* __restrict__ qg,
    const unsigned short* __restrict__ kg,
    const unsigned short* __restrict__ vg,
    const float* __restrict__ biasg,        // [H][N][N]
    const int* __restrict__ maskg,          // [S][N]
    unsigned short* __restrict__ ctx) {     // [S][N][C] bf16
  __shared__ unsigned short kls[N_ * DH_];  // 16 KB
  __shared__ unsigned short vls[N_ * DH_];  // 16 KB
  __shared__ unsigned short pls[4][16 * N_];// 32 KB (per-wave P staging)

  int tid = threadIdx.x, lane = tid & 31, w = tid >> 5;
  int bx = blockIdx.x;
  int s = bx >> 3, h = bx & 7;
  size_t base = (size_t)bx * N_ * DH_;

  // async DMA K and V into LDS: 1024 b128 transfers each, 8 per thread
#pragma unroll
  for (int i = 0; i < 8; ++i) {
    int t = tid + i * 128;                  // 0..1023
    async_ld_b128(kls + t * 8, kg + base + t * 8);
    async_ld_b128(vls + t * 8, vg + base + t * 8);
  }
  wait_async0();
  __syncthreads();

  int qrow0 = blockIdx.y * 64 + w * 16;

  // Q A-fragment gathered straight from global (K = DH = 32 -> single frag)
  v16u aq;
  {
    int m  = lane & 15;
    int kb = (lane & 16) ? 8 : 0;
    const unsigned short* qp = qg + base + (size_t)qrow0 * DH_;
#pragma unroll
    for (int i = 0; i < 16; ++i) {
      int k = kb + (i & 7) + ((i & 8) << 1);
      aq[i] = qp[m * DH_ + k];
    }
  }

  // scores = q @ k^T : 16 column tiles of 16
  v8f sc[16];
#pragma unroll
  for (int t = 0; t < 16; ++t) {
    v16u b = load_b_frag_nk(kls, t * 16, DH_, lane);
    sc[t] = wmma_bf16f32(aq, b, v8f_zero());
  }

  int hi = lane >> 4, nl = lane & 15;
  const int* mrow = maskg + s * N_;
  int km[16];
#pragma unroll
  for (int t = 0; t < 16; ++t) km[t] = mrow[t * 16 + nl];

#pragma unroll
  for (int r = 0; r < 8; ++r) {
    int qrow = qrow0 + r + 8 * hi;
    int qm = mrow[qrow];
    const float* brow = biasg + ((size_t)h * N_ + qrow) * N_;
    float mx = -3.4e38f;
#pragma unroll
    for (int t = 0; t < 16; ++t) {
      float v = sc[t][r];
      v = (qm != 0 && km[t] != 0) ? v * SCALE_ : NEG_;
      v += brow[t * 16 + nl];
      sc[t][r] = v;
      mx = fmaxf(mx, v);
    }
    // reduce across the 16 lanes holding this row (xor masks < 16 stay in half)
    mx = fmaxf(mx, __shfl_xor(mx, 8, 32));
    mx = fmaxf(mx, __shfl_xor(mx, 4, 32));
    mx = fmaxf(mx, __shfl_xor(mx, 2, 32));
    mx = fmaxf(mx, __shfl_xor(mx, 1, 32));
    float sum = 0.f;
#pragma unroll
    for (int t = 0; t < 16; ++t) {
      float e = __expf(sc[t][r] - mx);
      sc[t][r] = e;
      sum += e;
    }
    sum += __shfl_xor(sum, 8, 32);
    sum += __shfl_xor(sum, 4, 32);
    sum += __shfl_xor(sum, 2, 32);
    sum += __shfl_xor(sum, 1, 32);
    float inv = 1.f / sum;
    int prow = r + 8 * hi;
#pragma unroll
    for (int t = 0; t < 16; ++t)
      pls[w][prow * N_ + t * 16 + nl] = f32_to_bf16(sc[t][r] * inv);
  }

  // out = P @ V  (K = 256 keys in 8 steps of 32; DH = 32 -> 2 column tiles)
  v8f o0 = v8f_zero(), o1 = v8f_zero();
#pragma unroll
  for (int kk = 0; kk < 8; ++kk) {
    v16u a = load_a_frag(pls[w], 0, N_, kk * 32, lane);
    o0 = wmma_bf16f32(a, load_b_frag_kn(vls, kk * 32, 0, DH_, lane), o0);
    o1 = wmma_bf16f32(a, load_b_frag_kn(vls, kk * 32, 16, DH_, lane), o1);
  }

#pragma unroll
  for (int r = 0; r < 8; ++r) {
    int n = qrow0 + r + 8 * hi;
    size_t o = ((size_t)(s * N_ + n)) * C_ + h * DH_;
    ctx[o + nl]      = f32_to_bf16(o0[r]);
    ctx[o + 16 + nl] = f32_to_bf16(o1[r]);
  }
}

// ---------------------------------------------------------------------------
// K5: out projection + bias + residual  -> d_out (f32), same double-buffered
//     async-DMA GEMM structure as K3
// ---------------------------------------------------------------------------
__global__ __launch_bounds__(256) void outproj_gemm_kernel(
    const unsigned short* __restrict__ A,   // ctx bf16 [SN][256]
    const unsigned short* __restrict__ W,   // out_w bf16 [256][256]
    const float* __restrict__ bvec,         // [256]
    const float* __restrict__ msa,          // residual f32
    float* __restrict__ y) {
  __shared__ unsigned short at[2][128 * 32];
  __shared__ unsigned short bt[2][128 * 32];
  int tid = threadIdx.x, lane = tid & 31, w = tid >> 5;
  int row0 = blockIdx.x * 128;
  int col0 = blockIdx.y * 128;
  int wm = w >> 1, wn = w & 1;

  v8f acc[2][4];
#pragma unroll
  for (int mi = 0; mi < 2; ++mi)
#pragma unroll
    for (int j = 0; j < 4; ++j) acc[mi][j] = v8f_zero();

  async_tile_load(at[0], A, row0, 0, tid);
  async_tile_load(bt[0], W, col0, 0, tid);
  wait_async0();
  __syncthreads();

#pragma unroll
  for (int kk = 0; kk < 8; ++kk) {
    int cur = kk & 1;
    if (kk < 7) {
      async_tile_load(at[cur ^ 1], A, row0, (kk + 1) * 32, tid);
      async_tile_load(bt[cur ^ 1], W, col0, (kk + 1) * 32, tid);
    }
    v16u a0 = load_a_frag(at[cur], wm * 32, 32, 0, lane);
    v16u a1 = load_a_frag(at[cur], wm * 32 + 16, 32, 0, lane);
#pragma unroll
    for (int j = 0; j < 4; ++j) {
      v16u b = load_b_frag_nk(bt[cur], wn * 64 + j * 16, 32, lane);
      acc[0][j] = wmma_bf16f32(a0, b, acc[0][j]);
      acc[1][j] = wmma_bf16f32(a1, b, acc[1][j]);
    }
    wait_async0();
    __syncthreads();
  }

  int hi = lane >> 4, nl = lane & 15;
#pragma unroll
  for (int mi = 0; mi < 2; ++mi)
#pragma unroll
    for (int j = 0; j < 4; ++j)
#pragma unroll
      for (int r = 0; r < 8; ++r) {
        int grow = row0 + wm * 32 + mi * 16 + r + 8 * hi;
        int gcol = col0 + wn * 64 + j * 16 + nl;
        size_t o = (size_t)grow * C_ + gcol;
        y[o] = acc[mi][j][r] + bvec[gcol] + msa[o];
      }
}

// ---------------------------------------------------------------------------
// K6: per-row LayerNorm in place on d_out
// ---------------------------------------------------------------------------
__global__ __launch_bounds__(256) void ln_kernel(float* __restrict__ y,
                                                 const float* __restrict__ gamma,
                                                 const float* __restrict__ beta) {
  __shared__ float red[256];
  int row = blockIdx.x, t = threadIdx.x;
  size_t o = (size_t)row * C_ + t;
  float x = y[o];
  red[t] = x;
  __syncthreads();
  for (int off = 128; off > 0; off >>= 1) {
    if (t < off) red[t] += red[t + off];
    __syncthreads();
  }
  float mu = red[0] * (1.f / C_);
  __syncthreads();
  float d = x - mu;
  red[t] = d * d;
  __syncthreads();
  for (int off = 128; off > 0; off >>= 1) {
    if (t < off) red[t] += red[t + off];
    __syncthreads();
  }
  float var = red[0] * (1.f / C_);
  y[o] = d * rsqrtf(var + EPS_) * gamma[t] + beta[t];
}

// ---------------------------------------------------------------------------
// Workspace layout (bytes)
// ---------------------------------------------------------------------------
static constexpr size_t OFF_MSA_BF  = 0;          // SN*C bf16  = 16,777,216
static constexpr size_t OFF_CTX_BF  = 16777216;   // SN*C bf16
static constexpr size_t OFF_WQKV_BF = 33554432;   // 768*256 bf16 = 393,216
static constexpr size_t OFF_WOUT_BF = 33947648;   // 256*256 bf16 = 131,072
static constexpr size_t OFF_Q_BF    = 34078720;   // S*H*N*DH bf16 = 16,777,216
static constexpr size_t OFF_K_BF    = 50855936;
static constexpr size_t OFF_V_BF    = 67633152;
static constexpr size_t OFF_BIAS    = 84410368;   // H*N*N f32 = 2,097,152

extern "C" void kernel_launch(void* const* d_in, const int* in_sizes, int n_in,
                              void* d_out, int out_size, void* d_ws, size_t ws_size,
                              hipStream_t stream) {
  const float* msa    = (const float*)d_in[0];
  const float* pair   = (const float*)d_in[1];
  const int*   mask   = (const int*)d_in[2];
  const float* qkv_w  = (const float*)d_in[3];
  const float* qkv_b  = (const float*)d_in[4];
  const float* out_w  = (const float*)d_in[5];
  const float* out_b  = (const float*)d_in[6];
  const float* pb_w   = (const float*)d_in[7];
  const float* pb_b   = (const float*)d_in[8];
  const float* gamma  = (const float*)d_in[9];
  const float* beta   = (const float*)d_in[10];
  float* out = (float*)d_out;

  char* ws = (char*)d_ws;
  unsigned short* msa_bf  = (unsigned short*)(ws + OFF_MSA_BF);
  unsigned short* ctx_bf  = (unsigned short*)(ws + OFF_CTX_BF);
  unsigned short* wqkv_bf = (unsigned short*)(ws + OFF_WQKV_BF);
  unsigned short* wout_bf = (unsigned short*)(ws + OFF_WOUT_BF);
  unsigned short* q_bf    = (unsigned short*)(ws + OFF_Q_BF);
  unsigned short* k_bf    = (unsigned short*)(ws + OFF_K_BF);
  unsigned short* v_bf    = (unsigned short*)(ws + OFF_V_BF);
  float*          biasws  = (float*)(ws + OFF_BIAS);

  // K1: conversions
  f2bf_kernel<<<(SN_ * C_ + 255) / 256, 256, 0, stream>>>(msa, msa_bf, SN_ * C_);
  f2bf_kernel<<<(3 * C_ * C_ + 255) / 256, 256, 0, stream>>>(qkv_w, wqkv_bf, 3 * C_ * C_);
  f2bf_kernel<<<(C_ * C_ + 255) / 256, 256, 0, stream>>>(out_w, wout_bf, C_ * C_);

  // K2: pair bias
  pair_bias_kernel<<<(N_ * N_ + 255) / 256, 256, 0, stream>>>(pair, pb_w, pb_b, biasws);

  // K3: QKV projection (32768 x 768, K=256)
  qkv_gemm_kernel<<<dim3(SN_ / 128, 768 / 128), 256, 0, stream>>>(
      msa_bf, wqkv_bf, qkv_b, q_bf, k_bf, v_bf);

  // K4: fused masked-bias softmax attention
  attn_kernel<<<dim3(S_ * H_, 4), 128, 0, stream>>>(
      q_bf, k_bf, v_bf, biasws, mask, ctx_bf);

  // K5: out projection + bias + residual
  outproj_gemm_kernel<<<dim3(SN_ / 128, C_ / 128), 256, 0, stream>>>(
      ctx_bf, wout_bf, out_b, msa, out);

  // K6: layernorm
  ln_kernel<<<SN_, 256, 0, stream>>>(out, gamma, beta);
}
